// MultiheadSelfAttention_47931835023616
// MI455X (gfx1250) — matrix operand order repaired and checked by reading, not compiled, not run on hardware
//
#include <hip/hip_runtime.h>
#include <hip/hip_bf16.h>

// ---------------------------------------------------------------------------
// Multihead self-attention, B=2 T=2048 C=1024 H=16 D=64, fp32 in/out.
// Compute-bound => everything through V_WMMA_F32_16X16X32_BF16 (f32 accum).
// ---------------------------------------------------------------------------

#define Bn   2
#define Tn   2048
#define Cn   1024
#define Hn   16
#define Dn   64
#define Mn   (Bn * Tn)          // 4096 rows of the token matrix

typedef __attribute__((ext_vector_type(16))) __bf16 v16bf;
typedef __attribute__((ext_vector_type(8)))  __bf16 v8bf;
typedef __attribute__((ext_vector_type(8)))  float  v8f;

static __device__ __forceinline__ v16bf combine16(v8bf lo, v8bf hi) {
    v16bf r;
#pragma unroll
    for (int i = 0; i < 8; ++i) { r[i] = lo[i]; r[i + 8] = hi[i]; }
    return r;
}

static __device__ __forceinline__ v8f wmma_bf16(v16bf a, v16bf b, v8f c) {
    // 8 args: (neg_a, A, neg_b, B, c_mod, C, reuse_a, reuse_b)
    return __builtin_amdgcn_wmma_f32_16x16x32_bf16(false, a, false, b,
                                                   (short)0, c, false, false);
}

// ---------------------------------------------------------------------------
// Prep kernels: fp32 -> bf16 convert; weight transpose+convert (Wt[k][n]=W[n][k])
// ---------------------------------------------------------------------------
__global__ __launch_bounds__(256) void convert_bf16_kernel(
    const float* __restrict__ src, __bf16* __restrict__ dst, int n) {
    int i = blockIdx.x * 256 + threadIdx.x;
    if (i < n) dst[i] = (__bf16)src[i];
}

__global__ __launch_bounds__(1024) void transpose_w_kernel(
    const float* __restrict__ W, __bf16* __restrict__ Wt) {
    __shared__ float tile[32][33];
    const int tx = threadIdx.x, ty = threadIdx.y;
    // W is [N][K] (out_features x in_features); we want Wt[k][n] = W[n][k]
    tile[ty][tx] = W[(size_t)(blockIdx.y * 32 + ty) * Cn + blockIdx.x * 32 + tx];
    __syncthreads();
    Wt[(size_t)(blockIdx.x * 32 + ty) * Cn + blockIdx.y * 32 + tx] =
        (__bf16)tile[tx][ty];
}

// ---------------------------------------------------------------------------
// GEMM: out[m][n] = sum_k A[m][k] * Bt[k][n] + bias[n]
//   A  : [Mn][Cn] bf16 row-major
//   Bt : [Cn][Cn] bf16 row-major (pre-transposed weight)
// MODE 0: bf16 out, head-split [bh][t][d]      (Q / V)
// MODE 1: bf16 out, head-split [bh][d][t]      (K, pre-transposed for QK^T)
// MODE 2: f32  out, row-major  [m][n]          (final projection)
// Each wave computes a 32x64 tile: 2 A-fragments x 4 B-fragments -> 8 WMMAs
// per 32-wide K step (8 WMMA : 6 fragment loads).  8 waves -> 128x128 block.
// ---------------------------------------------------------------------------
template <int MODE>
__global__ __launch_bounds__(256) void gemm_bf16_kernel(
    const __bf16* __restrict__ A, const __bf16* __restrict__ Bt,
    const float* __restrict__ bias, __bf16* __restrict__ outB,
    float* __restrict__ outF) {
    const int lane = threadIdx.x & 31;
    const int wave = threadIdx.x >> 5;
    const int wm = wave & 3, wn = wave >> 2;
    const int m0 = blockIdx.y * 128 + wm * 32;     // 32 rows per wave
    const int n0 = blockIdx.x * 128 + wn * 64;     // 64 cols per wave
    const int rlane = lane & 15;
    const int hi = lane >> 4;

    v8f acc[2][4];
#pragma unroll
    for (int s = 0; s < 2; ++s)
#pragma unroll
        for (int j = 0; j < 4; ++j) acc[s][j] = v8f{};

    // A fragments: rows m0+rlane and m0+16+rlane; K chunks at +8*hi, +16
    const __bf16* arow0 = A + (size_t)(m0 + rlane) * Cn + hi * 8;
    const __bf16* arow1 = arow0 + (size_t)16 * Cn;

    for (int k0 = 0; k0 < Cn; k0 += 32) {
        // prefetch next K chunk (global_prefetch_b8)
        if (k0 + 32 < Cn) {
            __builtin_prefetch(arow0 + k0 + 32, 0, 3);
            __builtin_prefetch(Bt + (size_t)(k0 + 32 + lane) * Cn + n0, 0, 3);
        }
        v16bf afrag0 = combine16(*(const v8bf*)(arow0 + k0),
                                 *(const v8bf*)(arow0 + k0 + 16));
        v16bf afrag1 = combine16(*(const v8bf*)(arow1 + k0),
                                 *(const v8bf*)(arow1 + k0 + 16));
        // B fragment: column n0+j*16+rlane, K rows at +8*hi, +16 (matches A)
        const __bf16* brow = Bt + (size_t)(k0 + hi * 8) * Cn + n0 + rlane;
#pragma unroll
        for (int j = 0; j < 4; ++j) {
            v16bf bfrag;
#pragma unroll
            for (int e = 0; e < 16; ++e)
                bfrag[e] = brow[(size_t)(e < 8 ? e : e + 8) * Cn + j * 16];
            acc[0][j] = wmma_bf16(afrag0, bfrag, acc[0][j]);
            acc[1][j] = wmma_bf16(afrag1, bfrag, acc[1][j]);
        }
    }

    // Epilogue: C layout row = r + 8*hi, col = lane&15 within each 16x16 tile
#pragma unroll
    for (int s = 0; s < 2; ++s) {
#pragma unroll
        for (int j = 0; j < 4; ++j) {
            const int n = n0 + j * 16 + rlane;
            const float bv = bias[n];
            const int h = n >> 6, d = n & 63;
#pragma unroll
            for (int r = 0; r < 8; ++r) {
                const int m = m0 + s * 16 + r + 8 * hi;
                const float val = acc[s][j][r] + bv;
                if (MODE == 2) {
                    outF[(size_t)m * Cn + n] = val;
                } else {
                    const int bb = m >> 11, t = m & (Tn - 1);
                    const int bh = bb * Hn + h;
                    size_t off = (MODE == 0)
                                     ? ((size_t)bh * Tn + t) * Dn + d
                                     : ((size_t)bh * Dn + d) * Tn + t;
                    outB[off] = (__bf16)val;
                }
            }
        }
    }
}

// ---------------------------------------------------------------------------
// Flash attention (causal), one wave per (bh, 16-row Q block).
//   q  : [BH][T][D] bf16     kt : [BH][D][T] bf16     v : [BH][T][D] bf16
//   ao : [B][T][C] bf16 (c = h*64+d)  -- A matrix of the final projection
// Online softmax stats live per-lane: element (r,lane) of a C-tile is
// row (r + 8*(lane>=16)), col (lane&15); a 16-lane shfl_xor tree reduces rows.
// P goes through a per-wave LDS tile to re-shape C-layout -> A-fragment.
// ---------------------------------------------------------------------------
__global__ __launch_bounds__(128) void attn_kernel(
    const __bf16* __restrict__ q, const __bf16* __restrict__ kt,
    const __bf16* __restrict__ v, __bf16* __restrict__ ao) {
    __shared__ __bf16 Pb[4][16][32];

    const int lane = threadIdx.x & 31;
    const int wave = threadIdx.x >> 5;
    const int bh = blockIdx.y * 4 + wave;   // gridDim.y = BH/4 = 8
    const int t0 = blockIdx.x * 16;         // gridDim.x = T/16 = 128
    const int rlane = lane & 15;
    const int hi = lane >> 4;

    // Q A-fragments, K-dim = D = 64 -> two 16x32 fragments (loop invariant)
    const __bf16* qrow = q + ((size_t)bh * Tn + t0 + rlane) * Dn + hi * 8;
    const v16bf aq0 = combine16(*(const v8bf*)(qrow),      *(const v8bf*)(qrow + 16));
    const v16bf aq1 = combine16(*(const v8bf*)(qrow + 32), *(const v8bf*)(qrow + 48));

    v8f o[4];
#pragma unroll
    for (int j = 0; j < 4; ++j) o[j] = v8f{};
    float mr[8], lr[8];
#pragma unroll
    for (int r = 0; r < 8; ++r) { mr[r] = -__builtin_inff(); lr[r] = 0.0f; }

    const __bf16* kbase = kt + (size_t)bh * Dn * Tn;
    const __bf16* vbase0 = v + (size_t)bh * Tn * Dn;

    const int nblk = (t0 + 15) / 32 + 1;    // causal: only s0 <= t0+15
    for (int blk = 0; blk < nblk; ++blk) {
        const int s0 = blk * 32;

        // prefetch next key block's K and V rows (global_prefetch_b8)
        if (blk + 1 < nblk) {
            __builtin_prefetch(kbase + (size_t)lane * Tn + s0 + 32, 0, 3);
            __builtin_prefetch(vbase0 + (size_t)(s0 + 32 + lane) * Dn, 0, 3);
        }

        // ---- S = (Q K^T) * scale, two 16x16 column tiles, K-dim 64 ----
        v8f s[2];
#pragma unroll
        for (int jn = 0; jn < 2; ++jn) {
            s[jn] = v8f{};
            v16bf b0, b1;
#pragma unroll
            for (int e = 0; e < 16; ++e) {
                const int dk = hi * 8 + (e < 8 ? e : e + 8);
                b0[e] = kbase[(size_t)dk * Tn + s0 + jn * 16 + rlane];
                b1[e] = kbase[(size_t)(32 + dk) * Tn + s0 + jn * 16 + rlane];
            }
            s[jn] = wmma_bf16(aq0, b0, s[jn]);
            s[jn] = wmma_bf16(aq1, b1, s[jn]);
        }

        // ---- scale + causal mask ----
#pragma unroll
        for (int jn = 0; jn < 2; ++jn)
#pragma unroll
            for (int r = 0; r < 8; ++r) {
                const int scol = s0 + jn * 16 + rlane;
                const int trow = t0 + r + 8 * hi;
                const float sv = s[jn][r] * 0.125f;   // 1/sqrt(64)
                s[jn][r] = (scol > trow) ? -__builtin_inff() : sv;
            }

        // ---- online softmax: rowmax / exp / rowsum / rescale ----
        float corr[8];
#pragma unroll
        for (int r = 0; r < 8; ++r) {
            float rmax = fmaxf(s[0][r], s[1][r]);
#pragma unroll
            for (int msk = 8; msk >= 1; msk >>= 1)
                rmax = fmaxf(rmax, __shfl_xor(rmax, msk, 32));
            const float mn = fmaxf(mr[r], rmax);
            const float p0 = __expf(s[0][r] - mn);
            const float p1 = __expf(s[1][r] - mn);
            s[0][r] = p0; s[1][r] = p1;
            float rs = p0 + p1;
#pragma unroll
            for (int msk = 8; msk >= 1; msk >>= 1)
                rs += __shfl_xor(rs, msk, 32);
            corr[r] = __expf(mr[r] - mn);
            lr[r] = lr[r] * corr[r] + rs;
            mr[r] = mn;
        }

        // ---- P (C-layout) -> LDS -> A-fragment (per-wave buffer, in-order DS) ----
#pragma unroll
        for (int jn = 0; jn < 2; ++jn)
#pragma unroll
            for (int r = 0; r < 8; ++r)
                Pb[wave][r + 8 * hi][jn * 16 + rlane] = (__bf16)s[jn][r];
        const __bf16* prow = &Pb[wave][rlane][hi * 8];
        const v16bf ap = combine16(*(const v8bf*)(prow), *(const v8bf*)(prow + 16));

        // ---- O = O * corr + P V  (4 d-tiles) ----
#pragma unroll
        for (int j = 0; j < 4; ++j) {
#pragma unroll
            for (int r = 0; r < 8; ++r) o[j][r] *= corr[r];
        }
        const __bf16* vbase = vbase0 + (size_t)s0 * Dn;
#pragma unroll
        for (int j = 0; j < 4; ++j) {
            v16bf bv;
#pragma unroll
            for (int e = 0; e < 16; ++e)
                bv[e] = vbase[(size_t)(hi * 8 + (e < 8 ? e : e + 8)) * Dn + j * 16 + rlane];
            o[j] = wmma_bf16(ap, bv, o[j]);
        }
    }

    // ---- normalize + store head-merged bf16 [b][t][h*64+d] ----
    const int b = bh >> 4, h = bh & 15;
#pragma unroll
    for (int r = 0; r < 8; ++r) {
        const float inv = 1.0f / lr[r];
        const int t = t0 + r + 8 * hi;
        __bf16* orow = ao + ((size_t)b * Tn + t) * Cn + h * Dn;
#pragma unroll
        for (int j = 0; j < 4; ++j)
            orow[j * 16 + rlane] = (__bf16)(o[j][r] * inv);
    }
}

// ---------------------------------------------------------------------------
// Launch. Inputs: x, att_mask(unused: causal known), wq,bq, wk,bk, wv,bv, wp,bp
// Workspace map (bytes):
//   0        xb   [M][C]  bf16   8 MB
//   8  MB    wqt  [C][C]  bf16   2 MB
//   10 MB    wkt                 2 MB
//   12 MB    wvt                 2 MB
//   14 MB    wpt                 2 MB
//   16 MB    q    [BH][T][D]     8 MB
//   24 MB    kt   [BH][D][T]     8 MB
//   32 MB    v    [BH][T][D]     8 MB
//   40 MB    ao   [M][C]         8 MB          total 48 MB
// ---------------------------------------------------------------------------
extern "C" void kernel_launch(void* const* d_in, const int* in_sizes, int n_in,
                              void* d_out, int out_size, void* d_ws, size_t ws_size,
                              hipStream_t stream) {
    const float* x  = (const float*)d_in[0];
    const float* wq = (const float*)d_in[2];
    const float* bq = (const float*)d_in[3];
    const float* wk = (const float*)d_in[4];
    const float* bk = (const float*)d_in[5];
    const float* wv = (const float*)d_in[6];
    const float* bv = (const float*)d_in[7];
    const float* wp = (const float*)d_in[8];
    const float* bp = (const float*)d_in[9];
    float* out = (float*)d_out;

    char* ws = (char*)d_ws;
    const size_t MB = 1u << 20;
    __bf16* xb  = (__bf16*)(ws);
    __bf16* wqt = (__bf16*)(ws + 8 * MB);
    __bf16* wkt = (__bf16*)(ws + 10 * MB);
    __bf16* wvt = (__bf16*)(ws + 12 * MB);
    __bf16* wpt = (__bf16*)(ws + 14 * MB);
    __bf16* qh  = (__bf16*)(ws + 16 * MB);
    __bf16* kth = (__bf16*)(ws + 24 * MB);
    __bf16* vh  = (__bf16*)(ws + 32 * MB);
    __bf16* ao  = (__bf16*)(ws + 40 * MB);

    // 1) x -> bf16
    convert_bf16_kernel<<<(Mn * Cn) / 256, 256, 0, stream>>>(x, xb, Mn * Cn);
    // 2) weights -> transposed bf16 [K][N]
    dim3 tb(32, 32), tg(Cn / 32, Cn / 32);
    transpose_w_kernel<<<tg, tb, 0, stream>>>(wq, wqt);
    transpose_w_kernel<<<tg, tb, 0, stream>>>(wk, wkt);
    transpose_w_kernel<<<tg, tb, 0, stream>>>(wv, wvt);
    transpose_w_kernel<<<tg, tb, 0, stream>>>(wp, wpt);
    // 3) QKV projections (K stored pre-transposed per head)
    dim3 gg(Cn / 128, Mn / 128);
    gemm_bf16_kernel<0><<<gg, 256, 0, stream>>>(xb, wqt, bq, qh, nullptr);
    gemm_bf16_kernel<1><<<gg, 256, 0, stream>>>(xb, wkt, bk, kth, nullptr);
    gemm_bf16_kernel<0><<<gg, 256, 0, stream>>>(xb, wvt, bv, vh, nullptr);
    // 4) causal flash attention
    attn_kernel<<<dim3(Tn / 16, (Bn * Hn) / 4), 128, 0, stream>>>(qh, kth, vh, ao);
    // 5) output projection -> fp32
    gemm_bf16_kernel<2><<<gg, 256, 0, stream>>>(ao, wpt, bp, nullptr, out);
}
